// MultiBranchConditionalDiscriminator_47493748359862
// MI455X (gfx1250) — compile-verified, matrix operand order checked by reading, not run on hardware
//
#include <hip/hip_runtime.h>
#include <hip/hip_bf16.h>

// ---------------------------------------------------------------------------
// MultiBranchConditionalDiscriminator forward on gfx1250 (wave32, WMMA).
// conv1 direct -> f16 ; conv2 as im2col WMMA GEMM (async-to-LDS double
// buffered) with fused BN-stat accumulation ; BN+lrelu -> f16 ; 10-expert
// dense via WMMA (async double buffered) ; per-sample expert pick + stable
// class-order output.
// ---------------------------------------------------------------------------

typedef __attribute__((ext_vector_type(16))) _Float16 v16h;
typedef __attribute__((ext_vector_type(8)))  _Float16 v8h;
typedef __attribute__((ext_vector_type(8)))  float    v8f;
typedef __attribute__((ext_vector_type(4)))  int      v4i;

typedef __attribute__((address_space(1))) v4i gv4i;   // global
typedef __attribute__((address_space(3))) v4i lv4i;   // LDS

#define LRELU 0.2f
#define BN_EPS 1e-3f

static __device__ __forceinline__ float lrelu(float x) {
    return x >= 0.f ? x : LRELU * x;
}

// 16B async copy global -> LDS (GLOBAL_LOAD_ASYNC_TO_LDS_B128, ASYNCcnt).
static __device__ __forceinline__ void async_copy_b128(const _Float16* gsrc,
                                                       _Float16* ldst) {
    __builtin_amdgcn_global_load_async_to_lds_b128((gv4i*)gsrc, (lv4i*)ldst,
                                                   0, 0);
}

static __device__ __forceinline__ void wait_async0() {
#if __has_builtin(__builtin_amdgcn_s_wait_asynccnt)
    __builtin_amdgcn_s_wait_asynccnt(0);
#else
    asm volatile("s_wait_asynccnt 0x0" ::: "memory");
#endif
}

// Load one 16x32 f16 A/B fragment half-row from LDS.
// ISA layout (CDNA5 ISA): lanes 0-15 hold K=0..7 (VGPR0-3) and K=16..23
// (VGPR4-7); lanes 16-31 hold K=8..15 and K=24..31. Caller passes
// p = rowBase + (lane>=16 ? 8 : 0); we read p[0..7] and p[16..23].
static __device__ __forceinline__ v16h load_frag16(const _Float16* p) {
    v8h lo = *(const v8h*)(p);
    v8h hi = *(const v8h*)(p + 16);
    return __builtin_shufflevector(lo, hi, 0, 1, 2, 3, 4, 5, 6, 7,
                                           8, 9, 10, 11, 12, 13, 14, 15);
}

// ---------------------------------------------------------------------------
// Prep kernels: one-time weight convert / transpose, tiny.
// ---------------------------------------------------------------------------

// conv2_w f32 [1600][128] (k-major, HWIO flattened) -> f16 [128][1600] (n-major)
__global__ void prep_conv2w(const float* __restrict__ w, _Float16* __restrict__ wt) {
    int i = blockIdx.x * 256 + threadIdx.x;
    if (i >= 1600 * 128) return;
    int k = i / 128, n = i % 128;
    wt[(size_t)n * 1600 + k] = (_Float16)w[i];
}

// w1 f32 [10][6272][64] -> f16 [10][64][6272] (n-major per class)
__global__ void prep_w1(const float* __restrict__ w, _Float16* __restrict__ wt) {
    int i = blockIdx.x * 256 + threadIdx.x;
    if (i >= 10 * 6272 * 64) return;
    int c = i / (6272 * 64);
    int rem = i % (6272 * 64);
    int k = rem / 64, n = rem % 64;
    wt[((size_t)c * 64 + n) * 6272 + k] = (_Float16)w[i];
}

__global__ void zero_f32(float* __restrict__ p, int n) {
    int i = blockIdx.x * 256 + threadIdx.x;
    if (i < n) p[i] = 0.f;
}

// ---------------------------------------------------------------------------
// conv1: [4096,28,28,1] --5x5 s2 SAME--> lrelu -> f16 [4096,14,14,64]
// 4 output pixels per block; 64 threads (one per channel) per pixel.
// ---------------------------------------------------------------------------
__global__ __launch_bounds__(256) void conv1_kernel(
    const float* __restrict__ img, const float* __restrict__ w,
    const float* __restrict__ bias, _Float16* __restrict__ x1) {
    __shared__ float taps[4][25];
    const int g = threadIdx.x >> 6;     // pixel slot 0..3
    const int c = threadIdx.x & 63;     // channel
    const int pix = blockIdx.x * 4 + g; // < 802816
    const int b = pix / 196, rem = pix % 196;
    const int oy = rem / 14, ox = rem % 14;
    if (c < 25) {
        const int ky = c / 5, kx = c % 5;
        const int iy = oy * 2 - 1 + ky, ix = ox * 2 - 1 + kx;
        float v = 0.f;
        if (iy >= 0 && iy < 28 && ix >= 0 && ix < 28)
            v = img[((size_t)b * 28 + iy) * 28 + ix];
        taps[g][c] = v;
    }
    __syncthreads();
    float acc = bias[c];
#pragma unroll
    for (int t = 0; t < 25; ++t) acc += taps[g][t] * w[t * 64 + c];
    x1[(size_t)pix * 64 + c] = (_Float16)lrelu(acc);
}

// ---------------------------------------------------------------------------
// conv2 as WMMA GEMM: M=200704 (b*49+hw), K=1600 (ky,kx,ci), N=128.
// Block = 256 threads (8 waves) computes a 128x128 C tile.
// Double-buffered async-to-LDS K pipeline, 25 steps of K=64 (one conv tap
// each -> contiguous 64-ch f16 rows; border taps redirected to a zeroed
// global buffer). Fused BN sum / sumsq accumulation in the epilogue.
// Dynamic LDS: 2*A(128x72) + 2*B(128x72) halfs + 256 floats = 74752 B.
// ---------------------------------------------------------------------------
#define C2_TSTRIDE 72
#define C2_BUFH (128 * C2_TSTRIDE)

__global__ __launch_bounds__(256) void conv2_wmma(
    const _Float16* __restrict__ x1,   // [802816][64] f16
    const _Float16* __restrict__ wt,   // [128][1600]  f16 (n-major)
    const _Float16* __restrict__ zbuf, // [64] zeros
    const float* __restrict__ bias,    // [128]
    float* __restrict__ x2,            // [200704][128]
    float* __restrict__ sums) {        // [0..127]=sum, [128..255]=sumsq
    extern __shared__ _Float16 smem[];
    _Float16* AlBase = smem;                 // [2][C2_BUFH]
    _Float16* BlBase = smem + 2 * C2_BUFH;   // [2][C2_BUFH]
    float* sdsum = (float*)(smem + 4 * C2_BUFH);
    float* sdsq = sdsum + 128;

    const int tid = threadIdx.x;
    const int lane = tid & 31;
    const int wave = tid >> 5;
    const int mBase = blockIdx.x * 128;

    // loader role: each thread fills 32 halfs of one row (4 async b128)
    const int lrow = tid >> 1;
    const int lpart = (tid & 1) * 32;
    const int m = mBase + lrow;
    const int b = m / 49, rem = m % 49;
    const int oy = rem / 7, ox = rem % 7;

    if (tid < 128) { sdsum[tid] = 0.f; sdsq[tid] = 0.f; }

    v8f acc[8] = {};

    const int rsel = lane & 15;
    const int koff = (lane >= 16) ? 8 : 0;

    // ---- issue one K-slab of async fills into buffer `buf` ----
    auto issue = [&](int s, int buf) {
        const int ky = s / 5, kx = s % 5;
        const int iy = oy * 2 - 1 + ky, ix = ox * 2 - 1 + kx;
        const bool ok = (iy >= 0 && iy < 14 && ix >= 0 && ix < 14);
        const _Float16* asrc =
            ok ? x1 + (((size_t)(b * 14 + iy)) * 14 + ix) * 64 + lpart
               : zbuf + lpart;
        _Float16* ad = AlBase + buf * C2_BUFH + lrow * C2_TSTRIDE + lpart;
        const _Float16* bsrc = wt + (size_t)lrow * 1600 + s * 64 + lpart;
        _Float16* bd = BlBase + buf * C2_BUFH + lrow * C2_TSTRIDE + lpart;
#pragma unroll
        for (int q = 0; q < 4; ++q) {
            async_copy_b128(asrc + q * 8, ad + q * 8);
            async_copy_b128(bsrc + q * 8, bd + q * 8);
        }
    };

    issue(0, 0);
    wait_async0();
    __syncthreads();

    for (int s = 0; s < 25; ++s) {
        const int cur = s & 1;
        if (s + 1 < 25) issue(s + 1, cur ^ 1);

        const _Float16* aR =
            AlBase + cur * C2_BUFH + (wave * 16 + rsel) * C2_TSTRIDE + koff;
        const _Float16* bB = BlBase + cur * C2_BUFH;
        const v16h a0 = load_frag16(aR);
        const v16h a1 = load_frag16(aR + 32);
#pragma unroll
        for (int t = 0; t < 8; ++t) {
            const _Float16* bp = bB + (t * 16 + rsel) * C2_TSTRIDE + koff;
            const v16h b0 = load_frag16(bp);
            const v16h b1 = load_frag16(bp + 32);
            acc[t] = __builtin_amdgcn_wmma_f32_16x16x32_f16(
                false, a0, false, b0, (short)0, acc[t], false, false);
            acc[t] = __builtin_amdgcn_wmma_f32_16x16x32_f16(
                false, a1, false, b1, (short)0, acc[t], false, false);
        }
        wait_async0();
        __syncthreads();
    }

    // ---- epilogue: bias, store f32, fused BN partial sums ----
    const int rowBase = mBase + wave * 16 + ((lane >= 16) ? 8 : 0);
#pragma unroll
    for (int t = 0; t < 8; ++t) {
        const int ch = t * 16 + rsel;
        const float bv = bias[ch];
        float s1 = 0.f, s2 = 0.f;
#pragma unroll
        for (int r = 0; r < 8; ++r) {
            const float x = acc[t][r] + bv;
            x2[(size_t)(rowBase + r) * 128 + ch] = x;
            s1 += x;
            s2 += x * x;
        }
        atomicAdd(&sdsum[ch], s1);
        atomicAdd(&sdsq[ch], s2);
    }
    __syncthreads();
    if (tid < 128) {
        atomicAdd(&sums[tid], sdsum[tid]);
        atomicAdd(&sums[128 + tid], sdsq[tid]);
    }
}

// ---------------------------------------------------------------------------
// BN (batch stats) + lrelu + f16 convert: x2 [200704][128] -> xf16 [4096][6272]
// Flat layout already matches reshape(B,-1): i = b*6272 + hw*128 + ch.
// ---------------------------------------------------------------------------
__global__ __launch_bounds__(256) void bn_apply(
    const float* __restrict__ x2, const float* __restrict__ sums,
    const float* __restrict__ gamma, const float* __restrict__ beta,
    _Float16* __restrict__ xf16) {
    const int i = blockIdx.x * 256 + threadIdx.x;
    if (i >= 200704 * 128) return;
    const int ch = i & 127;
    const float invN = 1.f / 200704.f;
    const float mean = sums[ch] * invN;
    const float var = sums[128 + ch] * invN - mean * mean;
    const float scale = gamma[ch] * rsqrtf(var + BN_EPS);
    const float shift = beta[ch] - mean * scale;
    xf16[i] = (_Float16)lrelu(x2[i] * scale + shift);
}

// ---------------------------------------------------------------------------
// Expert layer 1 (all 10 experts dense): H[b][c][n] = lrelu(X[b,:]·W1[c][:,n]+b1)
// M=4096, K=6272, N=64. grid = (32 Mblocks, 10 classes), 128x64 tile / block.
// Double-buffered async-to-LDS K pipeline, 98 steps of K=64.
// ---------------------------------------------------------------------------
__global__ __launch_bounds__(256) void expert_gemm1(
    const _Float16* __restrict__ X,    // [4096][6272]
    const _Float16* __restrict__ W1t,  // [10][64][6272]
    const float* __restrict__ b1,      // [10][64]
    float* __restrict__ H) {           // [4096][10][64]
    __shared__ _Float16 Al[2][128 * C2_TSTRIDE];
    __shared__ _Float16 Bl[2][64 * C2_TSTRIDE];

    const int tid = threadIdx.x;
    const int lane = tid & 31;
    const int wave = tid >> 5;
    const int mBase = blockIdx.x * 128;
    const int cls = blockIdx.y;
    const _Float16* W = W1t + (size_t)cls * 64 * 6272;

    const int lrow = tid >> 1;
    const int lpart = (tid & 1) * 32;

    v8f acc[4] = {};
    const int rsel = lane & 15;
    const int koff = (lane >= 16) ? 8 : 0;
    const _Float16* xRow = X + (size_t)(mBase + lrow) * 6272 + lpart;
    const _Float16* wRow = W + (size_t)lrow * 6272 + lpart;

    auto issue = [&](int s, int buf) {
        const int k0 = s * 64;
        _Float16* ad = &Al[buf][lrow * C2_TSTRIDE + lpart];
#pragma unroll
        for (int q = 0; q < 4; ++q) async_copy_b128(xRow + k0 + q * 8, ad + q * 8);
        if (lrow < 64) {
            _Float16* bd = &Bl[buf][lrow * C2_TSTRIDE + lpart];
#pragma unroll
            for (int q = 0; q < 4; ++q) async_copy_b128(wRow + k0 + q * 8, bd + q * 8);
        }
    };

    issue(0, 0);
    wait_async0();
    __syncthreads();

    for (int s = 0; s < 98; ++s) {
        const int cur = s & 1;
        if (s + 1 < 98) issue(s + 1, cur ^ 1);

        const _Float16* aR = &Al[cur][(wave * 16 + rsel) * C2_TSTRIDE + koff];
        const v16h a0 = load_frag16(aR);
        const v16h a1 = load_frag16(aR + 32);
#pragma unroll
        for (int t = 0; t < 4; ++t) {
            const _Float16* bp = &Bl[cur][(t * 16 + rsel) * C2_TSTRIDE + koff];
            const v16h b0 = load_frag16(bp);
            const v16h b1 = load_frag16(bp + 32);
            acc[t] = __builtin_amdgcn_wmma_f32_16x16x32_f16(
                false, a0, false, b0, (short)0, acc[t], false, false);
            acc[t] = __builtin_amdgcn_wmma_f32_16x16x32_f16(
                false, a1, false, b1, (short)0, acc[t], false, false);
        }
        wait_async0();
        __syncthreads();
    }

    const int rowBase = mBase + wave * 16 + ((lane >= 16) ? 8 : 0);
#pragma unroll
    for (int t = 0; t < 4; ++t) {
        const int n = t * 16 + rsel;
        const float bv = b1[cls * 64 + n];
#pragma unroll
        for (int r = 0; r < 8; ++r) {
            const float x = lrelu(acc[t][r] + bv);
            H[((size_t)(rowBase + r) * 10 + cls) * 64 + n] = x;
        }
    }
}

// ---------------------------------------------------------------------------
// Expert layer 2 + per-sample class pick: val[i] = H[i][label[i]]·w2[c] + b2[c]
// ---------------------------------------------------------------------------
__global__ __launch_bounds__(256) void expert_out(
    const float* __restrict__ H, const float* __restrict__ w2,
    const float* __restrict__ b2, const int* __restrict__ labels,
    float* __restrict__ val) {
    const int i = blockIdx.x * 256 + threadIdx.x;
    if (i >= 4096) return;
    const int c = labels[i];
    const float* h = H + ((size_t)i * 10 + c) * 64;
    const float* w = w2 + c * 64;
    float s = b2[c];
#pragma unroll
    for (int k = 0; k < 64; ++k) s += h[k] * w[k];
    val[i] = s;
}

// ---------------------------------------------------------------------------
// Stable counting sort by label (10 classes) + scatter: out[pos[i]] = val[i].
// Single block; sequential stable-rank pass is 4096 iters (negligible).
// ---------------------------------------------------------------------------
__global__ __launch_bounds__(256) void order_gather(
    const int* __restrict__ labels, const float* __restrict__ val,
    float* __restrict__ out) {
    __shared__ int pos[4096];
    if (threadIdx.x == 0) {
        int cnt[10], off[10], seen[10];
        for (int c = 0; c < 10; ++c) { cnt[c] = 0; seen[c] = 0; }
        for (int i = 0; i < 4096; ++i) cnt[labels[i]]++;
        int run = 0;
        for (int c = 0; c < 10; ++c) { off[c] = run; run += cnt[c]; }
        for (int i = 0; i < 4096; ++i) {
            const int l = labels[i];
            pos[i] = off[l] + seen[l]++;
        }
    }
    __syncthreads();
    for (int i = threadIdx.x; i < 4096; i += 256) out[pos[i]] = val[i];
}

// ---------------------------------------------------------------------------
// Launch
// ---------------------------------------------------------------------------
static inline size_t alignUp(size_t x) { return (x + 255) & ~(size_t)255; }

extern "C" void kernel_launch(void* const* d_in, const int* in_sizes, int n_in,
                              void* d_out, int out_size, void* d_ws, size_t ws_size,
                              hipStream_t stream) {
    (void)in_sizes; (void)n_in; (void)out_size; (void)ws_size;

    const float* images  = (const float*)d_in[0];
    const int*   labels  = (const int*)d_in[1];
    const float* conv1_w = (const float*)d_in[2];
    const float* conv1_b = (const float*)d_in[3];
    const float* conv2_w = (const float*)d_in[4];
    const float* conv2_b = (const float*)d_in[5];
    const float* bn_g    = (const float*)d_in[6];
    const float* bn_b    = (const float*)d_in[7];
    const float* w1      = (const float*)d_in[8];
    const float* b1      = (const float*)d_in[9];
    const float* w2      = (const float*)d_in[10];
    const float* b2      = (const float*)d_in[11];
    float* out = (float*)d_out;

    char* ws = (char*)d_ws;
    size_t off = 0;
    _Float16* x1f16 = (_Float16*)(ws + off); off = alignUp(off + (size_t)802816 * 64 * 2);
    _Float16* w2t   = (_Float16*)(ws + off); off = alignUp(off + (size_t)128 * 1600 * 2);
    float*    x2f32 = (float*)   (ws + off); off = alignUp(off + (size_t)200704 * 128 * 4);
    float*    sums  = (float*)   (ws + off); off = alignUp(off + (256 + 32) * 4);
    _Float16* xf16  = (_Float16*)(ws + off); off = alignUp(off + (size_t)4096 * 6272 * 2);
    _Float16* w1t   = (_Float16*)(ws + off); off = alignUp(off + (size_t)10 * 64 * 6272 * 2);
    float*    Hbuf  = (float*)   (ws + off); off = alignUp(off + (size_t)4096 * 10 * 64 * 4);
    float*    val   = (float*)   (ws + off); off = alignUp(off + 4096 * 4);
    _Float16* zbuf  = (_Float16*)(sums + 256);  // 64 zero halfs after stats

    // weight prep + stat/zero-buffer zeroing
    prep_conv2w<<<(1600 * 128 + 255) / 256, 256, 0, stream>>>(conv2_w, w2t);
    prep_w1<<<(10 * 6272 * 64 + 255) / 256, 256, 0, stream>>>(w1, w1t);
    zero_f32<<<2, 256, 0, stream>>>(sums, 256 + 32);

    // conv1 -> f16
    conv1_kernel<<<802816 / 4, 256, 0, stream>>>(images, conv1_w, conv1_b, x1f16);

    // conv2 WMMA GEMM (async double-buffered) + fused BN stats
    const size_t c2_lds = (size_t)4 * C2_BUFH * sizeof(_Float16) + 256 * sizeof(float);
    conv2_wmma<<<200704 / 128, 256, c2_lds, stream>>>(x1f16, w2t, zbuf, conv2_b,
                                                      x2f32, sums);

    // BN apply + lrelu -> f16 activations
    bn_apply<<<(200704 * 128 + 255) / 256, 256, 0, stream>>>(x2f32, sums, bn_g, bn_b, xf16);

    // 10-expert dense layer via WMMA (async double-buffered)
    expert_gemm1<<<dim3(4096 / 128, 10), 256, 0, stream>>>(xf16, w1t, b1, Hbuf);

    // expert head + per-sample class pick
    expert_out<<<4096 / 256, 256, 0, stream>>>(Hbuf, w2, b2, labels, val);

    // stable class-order scatter
    order_gather<<<1, 256, 0, stream>>>(labels, val, out);
}
